// LSTM_21406117003944
// MI455X (gfx1250) — compile-verified, matrix operand order checked by reading, not run on hardware
//
#include <hip/hip_runtime.h>

// ---------- sizes ----------
#define VOCAB 256
#define EMBED 256
#define HIDDEN 1024
#define BATCH 128
#define SEQ 512
#define DIN (EMBED + HIDDEN)       // 1280
#define NGATES 4
#define NCOLS (NGATES * HIDDEN)    // 4096
#define KB_STEP (DIN / 32)         // 40 K-blocks of 32
#define NT_STEP (NCOLS / 16)       // 256 N-tiles of 16
#define KB_DEC (HIDDEN / 32)       // 32
#define NT_DEC (VOCAB / 16)        // 16

typedef __bf16 bf16x16 __attribute__((ext_vector_type(16)));
typedef float  f32x8   __attribute__((ext_vector_type(8)));

struct V32 { uint4 lo; uint4 hi; };   // 32 bytes == one v16bf fragment

__device__ __forceinline__ unsigned short f2bf(float f) {
    unsigned u = __float_as_uint(f);
    unsigned r = (u + 0x7FFFu + ((u >> 16) & 1u)) >> 16;   // RNE
    return (unsigned short)r;
}
__device__ __forceinline__ float sigf(float x) { return 1.0f / (1.0f + __expf(-x)); }

// ---------------------------------------------------------------------------
// Pack weights/emb/state into workspace.
// wpack layout: fragment (nt, kb) occupies 1024B at ((nt*KB+kb)*1024);
// within a fragment lane l owns 32B: 16 bf16 = K {k0..k0+7, k0+16..k0+23},
// k0 = (l>=16 ? 8 : 0), N = nt*16 + (l&15).   (ISA 7.12.2 16-bit B layout)
// ---------------------------------------------------------------------------
#define WP_ELEMS  (KB_STEP * NT_STEP * 512)   // 5,242,880 bf16
#define WD_ELEMS  (KB_DEC  * NT_DEC  * 512)   // 262,144
#define EMB_ELEMS (VOCAB * EMBED)             // 65,536
#define H_ELEMS   (BATCH * HIDDEN)            // 131,072

__global__ void pack_kernel(const float* __restrict__ Wf, const float* __restrict__ Wi,
                            const float* __restrict__ Wo, const float* __restrict__ Wc,
                            const float* __restrict__ Wd, const float* __restrict__ emb,
                            const float* __restrict__ h0, const float* __restrict__ c0,
                            unsigned short* __restrict__ wpack,
                            unsigned short* __restrict__ wdpack,
                            unsigned short* __restrict__ embBf,
                            unsigned short* __restrict__ hbuf0,
                            float* __restrict__ cbuf0) {
    const long long total = (long long)WP_ELEMS + WD_ELEMS + EMB_ELEMS + H_ELEMS + H_ELEMS;
    for (long long e = (long long)blockIdx.x * blockDim.x + threadIdx.x;
         e < total; e += (long long)gridDim.x * blockDim.x) {
        long long idx = e;
        if (idx < WP_ELEMS) {
            int frag = (int)(idx >> 9), w = (int)(idx & 511);
            int lane = w >> 4, el = w & 15;
            int chunk = el >> 3, pos = el & 7;
            int nt = frag / KB_STEP, kb = frag % KB_STEP;
            int N = nt * 16 + (lane & 15);
            int k = kb * 32 + ((lane >> 4) << 3) + chunk * 16 + pos;
            int gate = N >> 10, n = N & 1023;
            const float* W = (gate == 0) ? Wf : (gate == 1) ? Wi : (gate == 2) ? Wo : Wc;
            wpack[idx] = f2bf(W[(long long)k * HIDDEN + n]);
            continue;
        }
        idx -= WP_ELEMS;
        if (idx < WD_ELEMS) {
            int frag = (int)(idx >> 9), w = (int)(idx & 511);
            int lane = w >> 4, el = w & 15;
            int chunk = el >> 3, pos = el & 7;
            int nt = frag / KB_DEC, kb = frag % KB_DEC;
            int N = nt * 16 + (lane & 15);
            int k = kb * 32 + ((lane >> 4) << 3) + chunk * 16 + pos;
            wdpack[idx] = f2bf(Wd[(long long)k * VOCAB + N]);
            continue;
        }
        idx -= WD_ELEMS;
        if (idx < EMB_ELEMS) { embBf[idx] = f2bf(emb[idx]); continue; }
        idx -= EMB_ELEMS;
        if (idx < H_ELEMS) { hbuf0[idx] = f2bf(h0[idx]); continue; }
        idx -= H_ELEMS;
        cbuf0[idx] = c0[idx];
    }
}

// ---------------------------------------------------------------------------
// One LSTM timestep, with the decoder GEMM for the PREVIOUS step fused in
// (h_{t-1} is already staged in LDS as the K=[256,1280) slice of A).
// grid = (8 m-tiles, 16 n-groups), block = 256 (8 waves).
// Blocks with ng<2 additionally compute out[:, t-1, ng*128 : ng*128+128).
// ---------------------------------------------------------------------------
__global__ __launch_bounds__(256)
void lstm_step(const unsigned short* __restrict__ wpack,
               const unsigned short* __restrict__ wdpack,
               const unsigned short* __restrict__ embBf,
               const int* __restrict__ x,
               const unsigned short* __restrict__ hIn,
               unsigned short* __restrict__ hOut,
               const float* __restrict__ cIn,
               float* __restrict__ cOut,
               float* __restrict__ hF32,
               const float* __restrict__ bf_, const float* __restrict__ bi_,
               const float* __restrict__ bo_, const float* __restrict__ bc_,
               const float* __restrict__ bd_,
               float* __restrict__ out,
               int t) {
    __shared__ unsigned short As[16][1288];   // 16 x 1280 bf16, padded pitch
    __shared__ float gacc[4][16][68];         // f,i,o,g partial results

    const int tid = threadIdx.x;
    const int mt = blockIdx.x, ng = blockIdx.y;

    // ---- stage A = [emb(x_t) | h_{t-1}] rows into LDS (16B vector copies) ----
    for (int c = tid; c < 16 * 160; c += 256) {
        int r = c / 160, kc = (c % 160) * 8;
        int b = mt * 16 + r;
        const unsigned short* src;
        if (kc < EMBED) {
            int tok = x[b * SEQ + t];
            src = embBf + tok * EMBED + kc;
        } else {
            src = hIn + b * HIDDEN + (kc - EMBED);
        }
        *(uint4*)(&As[r][kc]) = *(const uint4*)src;
    }
    __syncthreads();

    // ---- WMMA GEMM: each wave -> one gate, two 16x16 N-subtiles ----
    const int wave = tid >> 5, lane = tid & 31;
    const int gate = wave >> 1, sp = wave & 1;
    const int nt0 = gate * 64 + ng * 4 + sp * 2;   // global 16-col tile index
    const int m  = lane & 15;
    const int k0 = (lane >> 4) << 3;               // element offset of first chunk

    f32x8 acc0 = {0.f,0.f,0.f,0.f,0.f,0.f,0.f,0.f};
    f32x8 acc1 = {0.f,0.f,0.f,0.f,0.f,0.f,0.f,0.f};
    const uint4* wq = (const uint4*)wpack;
    const unsigned short* arow = &As[m][0];

#pragma unroll 2
    for (int kb = 0; kb < KB_STEP; ++kb) {
        V32 av;
        av.lo = *(const uint4*)(arow + kb * 32 + k0);
        av.hi = *(const uint4*)(arow + kb * 32 + k0 + 16);
        bf16x16 afrag = __builtin_bit_cast(bf16x16, av);

        const uint4* p0 = wq + (((nt0 * KB_STEP + kb) << 6) + lane * 2);
        const uint4* p1 = wq + ((((nt0 + 1) * KB_STEP + kb) << 6) + lane * 2);
        __builtin_prefetch((const void*)(p0 + 64), 0, 0);  // next K-block, speculative
        __builtin_prefetch((const void*)(p1 + 64), 0, 0);

        V32 bv0, bv1;
        bv0.lo = p0[0]; bv0.hi = p0[1];
        bv1.lo = p1[0]; bv1.hi = p1[1];
        bf16x16 bfrag0 = __builtin_bit_cast(bf16x16, bv0);
        bf16x16 bfrag1 = __builtin_bit_cast(bf16x16, bv1);

        acc0 = __builtin_amdgcn_wmma_f32_16x16x32_bf16(false, afrag, false, bfrag0,
                                                       (short)0, acc0, false, false);
        acc1 = __builtin_amdgcn_wmma_f32_16x16x32_bf16(false, afrag, false, bfrag1,
                                                       (short)0, acc1, false, false);
    }

    // ---- scatter accumulators to LDS (C layout: VGPR j -> M=j / M=8+j) ----
#pragma unroll
    for (int j = 0; j < 8; ++j) {
        int mm = j + ((lane >> 4) << 3);
        gacc[gate][mm][(sp * 2 + 0) * 16 + (lane & 15)] = acc0[j];
        gacc[gate][mm][(sp * 2 + 1) * 16 + (lane & 15)] = acc1[j];
    }
    __syncthreads();

    // ---- gate combine: c' = f*c + i*g ; h' = o*tanh(c') ----
    for (int cell = tid; cell < 16 * 64; cell += 256) {
        int mm = cell >> 6, nn = cell & 63;
        int b = mt * 16 + mm;
        int n = ng * 64 + nn;
        float f = sigf(gacc[0][mm][nn] + bf_[n]);
        float i = sigf(gacc[1][mm][nn] + bi_[n]);
        float o = sigf(gacc[2][mm][nn] + bo_[n]);
        float g = tanhf(gacc[3][mm][nn] + bc_[n]);
        float cn = f * cIn[b * HIDDEN + n] + i * g;
        float hn = o * tanhf(cn);
        cOut[b * HIDDEN + n] = cn;
        hF32[b * HIDDEN + n] = hn;
        hOut[b * HIDDEN + n] = f2bf(hn);
    }

    // ---- fused decoder for step t-1: out[:,t-1,:] = h_{t-1} @ Wd + bd ----
    // Uses the K=[256,1280) slice of As (= h_{t-1}); no extra global h reads.
    if (ng < 2 && t > 0) {
        const int nt = ng * 8 + wave;              // 0..15 -> 256 vocab cols
        f32x8 dacc = {0.f,0.f,0.f,0.f,0.f,0.f,0.f,0.f};
        const uint4* dq = (const uint4*)wdpack;
        const unsigned short* hrow = &As[m][EMBED];

#pragma unroll 4
        for (int kb = 0; kb < KB_DEC; ++kb) {
            V32 av;
            av.lo = *(const uint4*)(hrow + kb * 32 + k0);
            av.hi = *(const uint4*)(hrow + kb * 32 + k0 + 16);
            bf16x16 afrag = __builtin_bit_cast(bf16x16, av);
            V32 bv;
            const uint4* p = dq + (((nt * KB_DEC + kb) << 6) + lane * 2);
            bv.lo = p[0]; bv.hi = p[1];
            bf16x16 bfrag = __builtin_bit_cast(bf16x16, bv);
            dacc = __builtin_amdgcn_wmma_f32_16x16x32_bf16(false, afrag, false, bfrag,
                                                           (short)0, dacc, false, false);
        }
        const int v = nt * 16 + (lane & 15);
        const float bias = bd_[v];
#pragma unroll
        for (int j = 0; j < 8; ++j) {
            int mm = j + ((lane >> 4) << 3);
            int b = mt * 16 + mm;
            out[((long long)b * SEQ + (t - 1)) * VOCAB + v] = dacc[j] + bias;
        }
    }
}

// ---------------------------------------------------------------------------
// Standalone decoder (only for the final timestep t = SEQ-1).
// grid = (8 m-tiles, 4 n-groups), block = 128 (4 waves, one 16x16 tile each).
// ---------------------------------------------------------------------------
__global__ __launch_bounds__(128)
void decoder_step(const unsigned short* __restrict__ wdpack,
                  const unsigned short* __restrict__ hbf,
                  const float* __restrict__ bd,
                  float* __restrict__ out, int t) {
    __shared__ unsigned short Hs[16][1032];

    const int tid = threadIdx.x;
    const int mt = blockIdx.x, ngd = blockIdx.y;

    for (int c = tid; c < 16 * 128; c += 128) {
        int r = c >> 7, kc = (c & 127) * 8;
        int b = mt * 16 + r;
        *(uint4*)(&Hs[r][kc]) = *(const uint4*)(hbf + b * HIDDEN + kc);
    }
    __syncthreads();

    const int wave = tid >> 5, lane = tid & 31;
    const int nt = ngd * 4 + wave;
    const int m  = lane & 15;
    const int k0 = (lane >> 4) << 3;

    f32x8 acc = {0.f,0.f,0.f,0.f,0.f,0.f,0.f,0.f};
    const uint4* wq = (const uint4*)wdpack;
    const unsigned short* arow = &Hs[m][0];

#pragma unroll 4
    for (int kb = 0; kb < KB_DEC; ++kb) {
        V32 av;
        av.lo = *(const uint4*)(arow + kb * 32 + k0);
        av.hi = *(const uint4*)(arow + kb * 32 + k0 + 16);
        bf16x16 afrag = __builtin_bit_cast(bf16x16, av);
        V32 bv;
        const uint4* p = wq + (((nt * KB_DEC + kb) << 6) + lane * 2);
        bv.lo = p[0]; bv.hi = p[1];
        bf16x16 bfrag = __builtin_bit_cast(bf16x16, bv);
        acc = __builtin_amdgcn_wmma_f32_16x16x32_bf16(false, afrag, false, bfrag,
                                                      (short)0, acc, false, false);
    }

    const int v = nt * 16 + (lane & 15);
    const float bias = bd[v];
#pragma unroll
    for (int j = 0; j < 8; ++j) {
        int mm = j + ((lane >> 4) << 3);
        int b = mt * 16 + mm;
        out[((long long)b * SEQ + t) * VOCAB + v] = acc[j] + bias;
    }
}

// final h, c -> tail of d_out
__global__ void tail_copy(const float* __restrict__ h, const float* __restrict__ c,
                          float* __restrict__ dst) {
    int i = blockIdx.x * blockDim.x + threadIdx.x;
    if (i < H_ELEMS) { dst[i] = h[i]; dst[H_ELEMS + i] = c[i]; }
}

// ---------------------------------------------------------------------------
extern "C" void kernel_launch(void* const* d_in, const int* in_sizes, int n_in,
                              void* d_out, int out_size, void* d_ws, size_t ws_size,
                              hipStream_t stream) {
    const int*   x   = (const int*)  d_in[0];
    const float* h0  = (const float*)d_in[1];
    const float* c0  = (const float*)d_in[2];
    const float* emb = (const float*)d_in[3];
    const float* Wf  = (const float*)d_in[4];  const float* bf_ = (const float*)d_in[5];
    const float* Wi  = (const float*)d_in[6];  const float* bi_ = (const float*)d_in[7];
    const float* Wo  = (const float*)d_in[8];  const float* bo_ = (const float*)d_in[9];
    const float* Wc  = (const float*)d_in[10]; const float* bc_ = (const float*)d_in[11];
    const float* Wd  = (const float*)d_in[12]; const float* bd_ = (const float*)d_in[13];
    float* out = (float*)d_out;

    char* ws = (char*)d_ws;
    size_t off = 0;
    unsigned short* wpack  = (unsigned short*)(ws + off); off += (size_t)WP_ELEMS * 2;
    unsigned short* wdpack = (unsigned short*)(ws + off); off += (size_t)WD_ELEMS * 2;
    unsigned short* embBf  = (unsigned short*)(ws + off); off += (size_t)EMB_ELEMS * 2;
    unsigned short* hbuf0  = (unsigned short*)(ws + off); off += (size_t)H_ELEMS * 2;
    unsigned short* hbuf1  = (unsigned short*)(ws + off); off += (size_t)H_ELEMS * 2;
    float* cbuf0 = (float*)(ws + off); off += (size_t)H_ELEMS * 4;
    float* cbuf1 = (float*)(ws + off); off += (size_t)H_ELEMS * 4;
    float* hF32  = (float*)(ws + off); off += (size_t)H_ELEMS * 4;

    pack_kernel<<<4096, 256, 0, stream>>>(Wf, Wi, Wo, Wc, Wd, emb, h0, c0,
                                          wpack, wdpack, embBf, hbuf0, cbuf0);

    for (int t = 0; t < SEQ; ++t) {
        const unsigned short* hin  = (t & 1) ? hbuf1 : hbuf0;
        unsigned short*       hout = (t & 1) ? hbuf0 : hbuf1;
        const float*          cin  = (t & 1) ? cbuf1 : cbuf0;
        float*                cout = (t & 1) ? cbuf0 : cbuf1;

        lstm_step<<<dim3(BATCH / 16, HIDDEN / 64), 256, 0, stream>>>(
            wpack, wdpack, embBf, x, hin, hout, cin, cout, hF32,
            bf_, bi_, bo_, bc_, bd_, out, t);
    }

    // decoder for the last timestep (h_{511} lives in hbuf0 after t=511)
    decoder_step<<<dim3(BATCH / 16, VOCAB / 64), 128, 0, stream>>>(
        wdpack, hbuf0, bd_, out, SEQ - 1);

    // after t=511 (odd): final c is in cbuf0; final h in hF32
    tail_copy<<<H_ELEMS / 256, 256, 0, stream>>>(hF32, cbuf0,
                                                 out + (size_t)BATCH * SEQ * VOCAB);
}
